// FantasyFootballLSTM_21809843929294
// MI455X (gfx1250) — matrix operand order repaired and checked by reading, not compiled, not run on hardware
//
#include <hip/hip_runtime.h>

typedef __bf16 bf16_t;
typedef __attribute__((ext_vector_type(16))) __bf16 v16bf;
typedef __attribute__((ext_vector_type(8)))  __bf16 v8bf;
typedef __attribute__((ext_vector_type(8)))  float  v8f;

// ---------------------------------------------------------------------------
// WMMA fragment loaders (layouts per CDNA5 ISA 7.12.2, wave32)
//   A 16x32 bf16: lane L -> row M=L&15; lanes<16 hold K {k0..k0+7, k0+16..23},
//                 lanes>=16 hold K {k0+8..15, k0+24..31}
//   B 32x16 bf16: lane L -> col N=n0+(L&15); K order same as A
//   C/D 16x16 f32: VGPR j: lanes<16 -> M=j, lanes>=16 -> M=8+j; N = lane&15
// ---------------------------------------------------------------------------
__device__ __forceinline__ v16bf frag_a(const bf16_t* base, int ld, int lane, int k0) {
    const int m   = lane & 15;
    const int kb0 = (lane & 16) ? 8 : 0;          // 0 or 8
    const bf16_t* r = base + (size_t)m * ld + k0;
    union { v16bf v; v8bf h[2]; } u;
    u.h[0] = *(const v8bf*)(r + kb0);             // K = kb0 .. kb0+7
    u.h[1] = *(const v8bf*)(r + kb0 + 16);        // K = kb0+16 .. kb0+23
    return u.v;
}

__device__ __forceinline__ v16bf frag_b(const bf16_t* base, int ld, int lane, int k0, int n0) {
    const int kb0 = (lane & 16) ? 8 : 0;
    const bf16_t* r = base + (size_t)(k0 + kb0) * ld + n0 + (lane & 15);
    v16bf v;
#pragma unroll
    for (int s = 0; s < 16; ++s) v[s] = r[(size_t)(s < 8 ? s : s + 8) * ld];
    return v;
}

__device__ __forceinline__ float sigm(float x) { return 1.0f / (1.0f + __expf(-x)); }

// ---------------------------------------------------------------------------
// fp32 -> bf16 conversion
// ---------------------------------------------------------------------------
__global__ void k_cvt(const float* __restrict__ in, bf16_t* __restrict__ out, int n) {
    int i = blockIdx.x * blockDim.x + threadIdx.x;
    if (i < n) out[i] = (bf16_t)in[i];
}

// ---------------------------------------------------------------------------
// Generic WMMA GEMM: C[M,N] = A_bf16[M,K] @ B_bf16[K,N] (+bias)(+relu)(+BN)(+res)
// 8 waves / block. Wave tile = 16x64 (4 accumulators): A fragment loaded once
// per K-step and reused for 4 WMMAs. Requires M%16==0, N%64==0, K%32==0.
// ---------------------------------------------------------------------------
__global__ void __launch_bounds__(256)
k_gemm(const bf16_t* __restrict__ A, const bf16_t* __restrict__ Bm,
       const float* __restrict__ bias, const float* __restrict__ res,
       const float* __restrict__ bn_g, const float* __restrict__ bn_b,
       const float* __restrict__ bn_m, const float* __restrict__ bn_v,
       float* __restrict__ Cf, bf16_t* __restrict__ Cbf,
       int M, int N, int K, int relu)
{
    const int lane = threadIdx.x & 31;
    const int wave = threadIdx.x >> 5;
    const int n64   = N >> 6;
    const int tiles = (M >> 4) * n64;
    const int t = blockIdx.x * 8 + wave;           // wave-uniform
    if (t >= tiles) return;
    const int m0 = (t / n64) << 4;
    const int n0 = (t % n64) << 6;

    v8f acc[4];
    {
        v8f z = {};
#pragma unroll
        for (int i = 0; i < 4; ++i) acc[i] = z;
    }

    const bf16_t* Abase = A + (size_t)m0 * K;
    const bf16_t* Arow  = A + (size_t)(m0 + (lane & 15)) * K;
    for (int k0 = 0; k0 < K; k0 += 32) {
        __builtin_prefetch(Arow + k0 + 64, 0, 1);
        v16bf a = frag_a(Abase, K, lane, k0);
#pragma unroll
        for (int i = 0; i < 4; ++i) {
            v16bf b = frag_b(Bm, N, lane, k0, n0 + 16 * i);
            acc[i] = __builtin_amdgcn_wmma_f32_16x16x32_bf16(false, a, false, b,
                                                             (short)0, acc[i], false, false);
        }
    }

#pragma unroll
    for (int i = 0; i < 4; ++i) {
        const int col = n0 + 16 * i + (lane & 15);
        const float bv = bias ? bias[col] : 0.0f;
        float g = 1.0f, be = 0.0f, mu = 0.0f, iv = 1.0f;
        if (bn_g) { g = bn_g[col]; be = bn_b[col]; mu = bn_m[col];
                    iv = rsqrtf(bn_v[col] + 1e-3f); }
#pragma unroll
        for (int j = 0; j < 8; ++j) {
            const int row = m0 + ((lane < 16) ? j : j + 8);
            float y = acc[i][j] + bv;
            if (relu) y = fmaxf(y, 0.0f);
            if (bn_g) y = g * (y - mu) * iv + be;
            if (res)  y += res[(size_t)row * N + col];
            if (Cf)  Cf [(size_t)row * N + col] = y;
            if (Cbf) Cbf[(size_t)row * N + col] = (bf16_t)y;
        }
    }
}

// ---------------------------------------------------------------------------
// Fused LSTM recurrence. Block = 16 batch rows, 8 waves.
// U[H,4H] staged in LDS once, reused for all T steps. Each wave owns
// NTW = G/128 contiguous 16-col tiles; the h fragment (LDS) is loaded once
// per K-step and reused for NTW WMMAs.
// Zx[B,T,4H] holds x@W + b (precomputed by k_gemm).
// ---------------------------------------------------------------------------
template<int H>
__global__ void __launch_bounds__(256)
k_lstm(const float* __restrict__ Zx, const bf16_t* __restrict__ U,
       float* __restrict__ seq_f, bf16_t* __restrict__ seq_bf,
       bf16_t* __restrict__ last_bf, int T)
{
    constexpr int G   = 4 * H;
    constexpr int NTW = G / 128;                 // 16-col tiles per wave (4/2/1)
    extern __shared__ char smem[];
    bf16_t* Ul = (bf16_t*)smem;                                   // H*G bf16
    bf16_t* hb = (bf16_t*)(smem + (size_t)H * G * 2);             // 16*H bf16
    float*  cb = (float*)(smem + (size_t)H * G * 2 + 16 * H * 2); // 16*H f32
    float*  zb = cb + 16 * H;                                     // 16*G f32

    const int tid  = threadIdx.x;
    const int lane = tid & 31;
    const int wave = tid >> 5;
    const int b0   = blockIdx.x * 16;
    const int nbase = wave * NTW * 16;           // this wave's column range

    for (int i = tid; i < H * G; i += 256) Ul[i] = U[i];
    for (int i = tid; i < 16 * H; i += 256) { hb[i] = (bf16_t)0.0f; cb[i] = 0.0f; }
    __syncthreads();

    for (int t = 0; t < T; ++t) {
        // z = Zx[:,t,:] + h @ U
        v8f acc[NTW];
#pragma unroll
        for (int i = 0; i < NTW; ++i) {
            const int col = nbase + 16 * i + (lane & 15);
#pragma unroll
            for (int j = 0; j < 8; ++j) {
                const int row = b0 + ((lane < 16) ? j : j + 8);
                acc[i][j] = Zx[((size_t)row * T + t) * G + col];
            }
        }
#pragma unroll
        for (int k0 = 0; k0 < H; k0 += 32) {
            v16bf a = frag_a(hb, H, lane, k0);
#pragma unroll
            for (int i = 0; i < NTW; ++i) {
                v16bf b = frag_b(Ul, G, lane, k0, nbase + 16 * i);
                acc[i] = __builtin_amdgcn_wmma_f32_16x16x32_bf16(false, a, false, b,
                                                                 (short)0, acc[i], false, false);
            }
        }
#pragma unroll
        for (int i = 0; i < NTW; ++i) {
            const int col = nbase + 16 * i + (lane & 15);
#pragma unroll
            for (int j = 0; j < 8; ++j) {
                const int r = (lane < 16) ? j : j + 8;
                zb[r * G + col] = acc[i][j];
            }
        }
        __syncthreads();

        // gates: i,f,g,o ; c = sig(f)*c + sig(i)*tanh(g) ; h = sig(o)*tanh(c)
        for (int i = tid; i < 16 * H; i += 256) {
            const int r = i / H, j = i % H;
            const float zi = zb[r * G + j];
            const float zf = zb[r * G + H + j];
            const float zg = zb[r * G + 2 * H + j];
            const float zo = zb[r * G + 3 * H + j];
            const float c  = sigm(zf) * cb[i] + sigm(zi) * tanhf(zg);
            const float h  = sigm(zo) * tanhf(c);
            cb[i] = c;
            hb[i] = (bf16_t)h;
            if (seq_f)  seq_f [((size_t)(b0 + r) * T + t) * H + j] = h;
            if (seq_bf) seq_bf[((size_t)(b0 + r) * T + t) * H + j] = (bf16_t)h;
        }
        __syncthreads();
    }

    if (last_bf) {
        for (int i = tid; i < 16 * H; i += 256) {
            const int r = i / H, j = i % H;
            last_bf[(size_t)(b0 + r) * H + j] = hb[i];
        }
    }
}

// ---------------------------------------------------------------------------
// Per-batch softmax attention: Q,K,V [B*T, NH*KD] f32, T=32, NH=4, KD=32.
// Block = one batch element, 128 threads = (head, query-row).
// ---------------------------------------------------------------------------
__global__ void __launch_bounds__(128)
k_attn(const float* __restrict__ Q, const float* __restrict__ K,
       const float* __restrict__ V, bf16_t* __restrict__ ctx_bf)
{
    __shared__ float Qs[32 * 128];
    __shared__ float Ks[32 * 128];
    __shared__ float Vs[32 * 128];
    const size_t base = (size_t)blockIdx.x * 32 * 128;
    for (int i = threadIdx.x; i < 32 * 128; i += 128) {
        Qs[i] = Q[base + i]; Ks[i] = K[base + i]; Vs[i] = V[base + i];
    }
    __syncthreads();

    const int hd = threadIdx.x >> 5;       // head
    const int tq = threadIdx.x & 31;       // query row
    const float* q = &Qs[tq * 128 + hd * 32];

    float s[32];
    float mx = -1e30f;
#pragma unroll
    for (int u = 0; u < 32; ++u) {
        const float* kk = &Ks[u * 128 + hd * 32];
        float d = 0.0f;
#pragma unroll
        for (int j = 0; j < 32; ++j) d += q[j] * kk[j];
        d *= 0.17677669529663687f;         // 1/sqrt(32)
        s[u] = d;
        mx = fmaxf(mx, d);
    }
    float sum = 0.0f;
#pragma unroll
    for (int u = 0; u < 32; ++u) { s[u] = __expf(s[u] - mx); sum += s[u]; }
    const float inv = 1.0f / sum;
#pragma unroll
    for (int j = 0; j < 32; ++j) {
        float acc = 0.0f;
#pragma unroll
        for (int u = 0; u < 32; ++u) acc += s[u] * Vs[u * 128 + hd * 32 + j];
        ctx_bf[base + tq * 128 + hd * 32 + j] = (bf16_t)(acc * inv);
    }
}

// ---------------------------------------------------------------------------
// Final projection [B,64] @ [64,1] + b
// ---------------------------------------------------------------------------
__global__ void k_outdot(const bf16_t* __restrict__ X, const float* __restrict__ w,
                         const float* __restrict__ b, float* __restrict__ out,
                         int M, int Kd)
{
    int r = blockIdx.x * blockDim.x + threadIdx.x;
    if (r >= M) return;
    float acc = b[0];
    for (int k = 0; k < Kd; ++k) acc += (float)X[(size_t)r * Kd + k] * w[k];
    out[r] = acc;
}

// ---------------------------------------------------------------------------
// Host side
// ---------------------------------------------------------------------------
extern "C" void kernel_launch(void* const* d_in, const int* in_sizes, int n_in,
                              void* d_out, int out_size, void* d_ws, size_t ws_size,
                              hipStream_t stream)
{
    (void)in_sizes; (void)n_in; (void)out_size; (void)ws_size;
    constexpr int  Bb = 8192, T = 32, F = 128;
    constexpr int  H1 = 128, H2 = 64, H3 = 32;
    constexpr int  D1 = 256, D2 = 128, D3 = 64;
    constexpr long BT = (long)Bb * T;

    const float* inp   = (const float*)d_in[0];
    const float* l1_W  = (const float*)d_in[1];
    const float* l1_U  = (const float*)d_in[2];
    const float* l1_b  = (const float*)d_in[3];
    const float* l2_W  = (const float*)d_in[4];
    const float* l2_U  = (const float*)d_in[5];
    const float* l2_b  = (const float*)d_in[6];
    const float* l3_W  = (const float*)d_in[7];
    const float* l3_U  = (const float*)d_in[8];
    const float* l3_b  = (const float*)d_in[9];
    const float* Wq    = (const float*)d_in[10];
    const float* bq    = (const float*)d_in[11];
    const float* Wk    = (const float*)d_in[12];
    const float* bk    = (const float*)d_in[13];
    const float* Wv    = (const float*)d_in[14];
    const float* bv    = (const float*)d_in[15];
    const float* Wo    = (const float*)d_in[16];
    const float* bo    = (const float*)d_in[17];
    const float* d1_w  = (const float*)d_in[18];
    const float* d1_b  = (const float*)d_in[19];
    const float* bn1g  = (const float*)d_in[20];
    const float* bn1b  = (const float*)d_in[21];
    const float* bn1m  = (const float*)d_in[22];
    const float* bn1v  = (const float*)d_in[23];
    const float* d2_w  = (const float*)d_in[24];
    const float* d2_b  = (const float*)d_in[25];
    const float* bn2g  = (const float*)d_in[26];
    const float* bn2b  = (const float*)d_in[27];
    const float* bn2m  = (const float*)d_in[28];
    const float* bn2v  = (const float*)d_in[29];
    const float* d3_w  = (const float*)d_in[30];
    const float* d3_b  = (const float*)d_in[31];
    const float* bn3g  = (const float*)d_in[32];
    const float* bn3b  = (const float*)d_in[33];
    const float* bn3m  = (const float*)d_in[34];
    const float* bn3v  = (const float*)d_in[35];
    const float* out_w = (const float*)d_in[36];
    const float* out_b = (const float*)d_in[37];

    // ---- carve workspace ----
    char* p = (char*)d_ws;
    auto carve = [&](size_t bytes) -> char* {
        char* r = p; p += (bytes + 255) & ~(size_t)255; return r;
    };
    bf16_t* Xbf   = (bf16_t*)carve(BT * F * 2);
    bf16_t* wl1W  = (bf16_t*)carve((size_t)F  * 4 * H1 * 2);
    bf16_t* wl1U  = (bf16_t*)carve((size_t)H1 * 4 * H1 * 2);
    bf16_t* wl2W  = (bf16_t*)carve((size_t)H1 * 4 * H2 * 2);
    bf16_t* wl2U  = (bf16_t*)carve((size_t)H2 * 4 * H2 * 2);
    bf16_t* wl3W  = (bf16_t*)carve((size_t)H2 * 4 * H3 * 2);
    bf16_t* wl3U  = (bf16_t*)carve((size_t)H3 * 4 * H3 * 2);
    bf16_t* wqb   = (bf16_t*)carve((size_t)H2 * 128 * 2);
    bf16_t* wkb   = (bf16_t*)carve((size_t)H2 * 128 * 2);
    bf16_t* wvb   = (bf16_t*)carve((size_t)H2 * 128 * 2);
    bf16_t* wob   = (bf16_t*)carve((size_t)128 * H2 * 2);
    bf16_t* wd1   = (bf16_t*)carve((size_t)H3 * D1 * 2);
    bf16_t* wd2   = (bf16_t*)carve((size_t)D1 * D2 * 2);
    bf16_t* wd3   = (bf16_t*)carve((size_t)D2 * D3 * 2);
    float*  Zx    = (float*) carve(BT * 512 * 4);   // reused for Zx1/Zx2/Zx3
    bf16_t* h1bf  = (bf16_t*)carve(BT * H1 * 2);
    float*  h2f   = (float*) carve(BT * H2 * 4);
    bf16_t* h2bf  = (bf16_t*)carve(BT * H2 * 2);
    float*  Qb    = (float*) carve(BT * 128 * 4);
    float*  Kb    = (float*) carve(BT * 128 * 4);
    float*  Vb    = (float*) carve(BT * 128 * 4);
    bf16_t* ctxbf = (bf16_t*)carve(BT * 128 * 2);
    bf16_t* x2bf  = (bf16_t*)carve(BT * H2 * 2);
    bf16_t* h3bf  = (bf16_t*)carve((size_t)Bb * H3 * 2);
    bf16_t* e1bf  = (bf16_t*)carve((size_t)Bb * D1 * 2);
    bf16_t* e2bf  = (bf16_t*)carve((size_t)Bb * D2 * 2);
    bf16_t* e3bf  = (bf16_t*)carve((size_t)Bb * D3 * 2);

    auto cvt = [&](const float* src, bf16_t* dst, long n) {
        k_cvt<<<(int)((n + 255) / 256), 256, 0, stream>>>(src, dst, (int)n);
    };
    auto gemm = [&](const bf16_t* A, const bf16_t* Bw, const float* bias,
                    const float* res, const float* g, const float* be,
                    const float* mu, const float* var,
                    float* Cf, bf16_t* Cbf, long M, int N, int Kk, int relu) {
        long tiles = (M / 16) * (N / 64);
        k_gemm<<<(int)((tiles + 7) / 8), 256, 0, stream>>>(
            A, Bw, bias, res, g, be, mu, var, Cf, Cbf, (int)M, N, Kk, relu);
    };

    // LDS budgets for the fused LSTM kernels (U + h + c + z)
    constexpr int SH1 = H1*4*H1*2 + 16*H1*2 + 16*H1*4 + 16*4*H1*4;   // 176128
    constexpr int SH2 = H2*4*H2*2 + 16*H2*2 + 16*H2*4 + 16*4*H2*4;   //  55296
    constexpr int SH3 = H3*4*H3*2 + 16*H3*2 + 16*H3*4 + 16*4*H3*4;   //  19456
    (void)hipFuncSetAttribute((const void*)k_lstm<H1>,
                              hipFuncAttributeMaxDynamicSharedMemorySize, SH1);
    (void)hipFuncSetAttribute((const void*)k_lstm<H2>,
                              hipFuncAttributeMaxDynamicSharedMemorySize, SH2);
    (void)hipFuncSetAttribute((const void*)k_lstm<H3>,
                              hipFuncAttributeMaxDynamicSharedMemorySize, SH3);

    // ---- convert activations/weights to bf16 ----
    cvt(inp,  Xbf,  BT * F);
    cvt(l1_W, wl1W, (long)F  * 4 * H1);  cvt(l1_U, wl1U, (long)H1 * 4 * H1);
    cvt(l2_W, wl2W, (long)H1 * 4 * H2);  cvt(l2_U, wl2U, (long)H2 * 4 * H2);
    cvt(l3_W, wl3W, (long)H2 * 4 * H3);  cvt(l3_U, wl3U, (long)H3 * 4 * H3);
    cvt(Wq, wqb, (long)H2 * 128); cvt(Wk, wkb, (long)H2 * 128);
    cvt(Wv, wvb, (long)H2 * 128); cvt(Wo, wob, (long)128 * H2);
    cvt(d1_w, wd1, (long)H3 * D1); cvt(d2_w, wd2, (long)D1 * D2);
    cvt(d3_w, wd3, (long)D2 * D3);

    // ---- LSTM 1 : [B,T,128] -> [B,T,128] ----
    gemm(Xbf, wl1W, l1_b, nullptr, nullptr, nullptr, nullptr, nullptr,
         Zx, nullptr, BT, 4 * H1, F, 0);
    k_lstm<H1><<<Bb / 16, 256, SH1, stream>>>(Zx, wl1U, nullptr, h1bf, nullptr, T);

    // ---- LSTM 2 : [B,T,128] -> [B,T,64] ----
    gemm(h1bf, wl2W, l2_b, nullptr, nullptr, nullptr, nullptr, nullptr,
         Zx, nullptr, BT, 4 * H2, H1, 0);
    k_lstm<H2><<<Bb / 16, 256, SH2, stream>>>(Zx, wl2U, h2f, h2bf, nullptr, T);

    // ---- MHA on [B,T,64], 4 heads x 32 ----
    gemm(h2bf, wqb, bq, nullptr, nullptr, nullptr, nullptr, nullptr,
         Qb, nullptr, BT, 128, H2, 0);
    gemm(h2bf, wkb, bk, nullptr, nullptr, nullptr, nullptr, nullptr,
         Kb, nullptr, BT, 128, H2, 0);
    gemm(h2bf, wvb, bv, nullptr, nullptr, nullptr, nullptr, nullptr,
         Vb, nullptr, BT, 128, H2, 0);
    k_attn<<<Bb, 128, 0, stream>>>(Qb, Kb, Vb, ctxbf);
    // out-proj + bias + residual (x + attended), store bf16 for next GEMM
    gemm(ctxbf, wob, bo, h2f, nullptr, nullptr, nullptr, nullptr,
         nullptr, x2bf, BT, H2, 128, 0);

    // ---- LSTM 3 : [B,T,64] -> last h [B,32] ----
    gemm(x2bf, wl3W, l3_b, nullptr, nullptr, nullptr, nullptr, nullptr,
         Zx, nullptr, BT, 4 * H3, H2, 0);
    k_lstm<H3><<<Bb / 16, 256, SH3, stream>>>(Zx, wl3U, nullptr, nullptr, h3bf, T);

    // ---- Dense + BN + ReLU chain ----
    gemm(h3bf, wd1, d1_b, nullptr, bn1g, bn1b, bn1m, bn1v,
         nullptr, e1bf, Bb, D1, H3, 1);
    gemm(e1bf, wd2, d2_b, nullptr, bn2g, bn2b, bn2m, bn2v,
         nullptr, e2bf, Bb, D2, D1, 1);
    gemm(e2bf, wd3, d3_b, nullptr, bn3g, bn3b, bn3m, bn3v,
         nullptr, e3bf, Bb, D3, D2, 1);

    // ---- Output head [B,64] -> [B,1] ----
    k_outdot<<<(Bb + 255) / 256, 256, 0, stream>>>(e3bf, out_w, out_b,
                                                   (float*)d_out, Bb, D3);
}
